// Sent_Posit_Drmm_Modeler_77300821393900
// MI455X (gfx1250) — matrix operand order repaired and checked by reading, not compiled, not run on hardware
//
#include <hip/hip_runtime.h>
#include <hip/hip_bf16.h>

// ---------------------------------------------------------------------------
// Sent_Posit_Drmm_Modeler on gfx1250 (MI455X).
// GEMM stages (conv1d-as-3-GEMMs + 4 cosine GEMMs) run on v_wmma_f32_16x16x32_f16
// (f16 operands, f32 accumulate). Working set is L2-resident (192MB), so WMMA
// fragments are loaded directly from global with aligned b128 loads following
// the ISA per-lane layouts (CDNA5 ISA 7.12.2). Each wave computes a
// 32x32 output tile (2x2 WMMA register blocking) -> 2 b128 loads per wmma.
// ---------------------------------------------------------------------------

typedef __attribute__((ext_vector_type(16))) _Float16 v16h;
typedef __attribute__((ext_vector_type(8)))  _Float16 v8h;
typedef __attribute__((ext_vector_type(8)))  float    v8f;

#define VE     300     // embedding dim E
#define QN     64      // query length
#define DN     4096    // doc length
#define KP     320     // E padded to multiple of 32 (WMMA K step)
#define NSLOPE 0.01f

// ---------------- WMMA fragment loaders (wave32, 16x16x32 f16) --------------
// A (MxK=16x32): lane l holds row m0+(l%16); lanes 0-15: K = k0+{0..7, 16..23},
// lanes 16-31: K = k0+{8..15, 24..31}.
__device__ __forceinline__ v16h load_a_frag(const _Float16* rowBase, int k0, bool hi) {
    const _Float16* p = rowBase + k0 + (hi ? 8 : 0);
    v8h x0 = *(const v8h*)(p);
    v8h x1 = *(const v8h*)(p + 16);
    v16h a;
#pragma unroll
    for (int j = 0; j < 8; ++j) { a[j] = x0[j]; a[j + 8] = x1[j]; }
    return a;
}
// B (KxN=32x16) sourced from a row-major [N][K] matrix (i.e. D = A * B^T):
// lane l holds column n0+(l%16); lanes 0-15: K = k0+{0..7, 16..23}, lanes 16-31: K = k0+{8..15, 24..31}.
__device__ __forceinline__ v16h load_b_frag(const _Float16* rowBase, int k0, bool hi) {
    const _Float16* p = rowBase + k0 + (hi ? 8 : 0);
    v8h x0 = *(const v8h*)(p);
    v8h x1 = *(const v8h*)(p + 16);
    v16h b;
#pragma unroll
    for (int j = 0; j < 8; ++j) { b[j] = x0[j]; b[j + 8] = x1[j]; }
    return b;
}

// ---------------- 1. gather + f32->f16 convert into padded buffers ----------
// dst layout: [L+2][KP] f16, row 0 and row L+1 zero (conv halo), cols >=300 zero.
__global__ void gather_kernel(const int* __restrict__ question,
                              const int* __restrict__ doc1,
                              const int* __restrict__ doc2,
                              const float* __restrict__ emb,
                              _Float16* __restrict__ qpad,
                              _Float16* __restrict__ d1pad,
                              _Float16* __restrict__ d2pad) {
    int r = blockIdx.x;
    int c = threadIdx.x;           // blockDim.x == KP
    _Float16* dst;
    const int* idx;
    int L, prow;
    if (r < QN + 2)                 { dst = qpad;  idx = question; L = QN; prow = r; }
    else if (r < QN + 2 + DN + 2)   { dst = d1pad; idx = doc1;     L = DN; prow = r - (QN + 2); }
    else                            { dst = d2pad; idx = doc2;     L = DN; prow = r - (QN + 2) - (DN + 2); }
    float v = 0.0f;
    if (prow > 0 && prow <= L && c < VE) {
        int vid = idx[prow - 1];
        v = emb[(size_t)vid * VE + c];
    }
    dst[(size_t)prow * KP + c] = (_Float16)v;
}

// ---------------- 2. conv taps: w_t[o][i] = conv_w[o][i][t], padded ---------
__global__ void taps_kernel(const float* __restrict__ conv_w,
                            _Float16* __restrict__ w0,
                            _Float16* __restrict__ w1,
                            _Float16* __restrict__ w2) {
    int t   = blockIdx.x / KP;     // 0..2
    int row = blockIdx.x % KP;     // output channel o
    int c   = threadIdx.x;         // input channel i
    float v = 0.0f;
    if (row < VE && c < VE) v = conv_w[((size_t)row * VE + c) * 3 + t];
    _Float16* w = (t == 0) ? w0 : ((t == 1) ? w1 : w2);
    w[(size_t)row * KP + c] = (_Float16)v;
}

// ---------------- 3. row norms: invn[r] = rsqrt(sum x^2) --------------------
__global__ void norm_kernel(const _Float16* __restrict__ src, int L,
                            float* __restrict__ invn) {
    int wave = threadIdx.x >> 5;
    int lane = threadIdx.x & 31;
    int row  = blockIdx.x * 4 + wave;
    if (row >= L) return;
    const _Float16* p = src + (size_t)row * KP;
    float s = 0.0f;
    for (int j = lane; j < KP; j += 32) { float f = (float)p[j]; s += f * f; }
#pragma unroll
    for (int m = 16; m > 0; m >>= 1) s += __shfl_xor(s, m);
    if (lane == 0) invn[row] = rsqrtf(s);
}

// ---------------- 4. conv1d residual via 3 shifted WMMA GEMMs ---------------
// y[m][n] = leaky( sum_t sum_k xpad[m+t][k] * w_t[n][k] + b[n] ) + x[m][n]
// 2x2 register blocking: each wave produces a 32x32 output tile.
__global__ __launch_bounds__(32)
void conv_kernel(const _Float16* __restrict__ xpad,
                 const _Float16* __restrict__ w0,
                 const _Float16* __restrict__ w1,
                 const _Float16* __restrict__ w2,
                 const float* __restrict__ bias,
                 _Float16* __restrict__ out, int L) {
    int m0 = blockIdx.x * 32;
    int n0 = blockIdx.y * 32;
    int lane = threadIdx.x;
    int mr = lane & 15;
    bool hi = lane >= 16;

    v8f acc00 = {}, acc01 = {}, acc10 = {}, acc11 = {};
#pragma unroll
    for (int t = 0; t < 3; ++t) {
        const _Float16* W = (t == 0) ? w0 : ((t == 1) ? w1 : w2);
        const _Float16* Arow0 = xpad + (size_t)(m0 + mr + t) * KP;       // X_t row = xpad[m+t]
        const _Float16* Arow1 = Arow0 + (size_t)16 * KP;
        const _Float16* Brow0 = W + (size_t)(n0 + mr) * KP;
        const _Float16* Brow1 = Brow0 + (size_t)16 * KP;
#pragma unroll
        for (int k0 = 0; k0 < KP; k0 += 32) {
            v16h a0 = load_a_frag(Arow0, k0, hi);
            v16h a1 = load_a_frag(Arow1, k0, hi);
            v16h b0 = load_b_frag(Brow0, k0, hi);
            v16h b1 = load_b_frag(Brow1, k0, hi);
            acc00 = __builtin_amdgcn_wmma_f32_16x16x32_f16(false, a0, false, b0, (short)0, acc00, false, false);
            acc01 = __builtin_amdgcn_wmma_f32_16x16x32_f16(false, a0, false, b1, (short)0, acc01, false, false);
            acc10 = __builtin_amdgcn_wmma_f32_16x16x32_f16(false, a1, false, b0, (short)0, acc10, false, false);
            acc11 = __builtin_amdgcn_wmma_f32_16x16x32_f16(false, a1, false, b1, (short)0, acc11, false, false);
        }
    }

#pragma unroll
    for (int ti = 0; ti < 2; ++ti) {
#pragma unroll
        for (int tj = 0; tj < 2; ++tj) {
            const v8f& acc = (ti == 0) ? ((tj == 0) ? acc00 : acc01)
                                       : ((tj == 0) ? acc10 : acc11);
            int n = n0 + tj * 16 + mr;
            int mBase = m0 + ti * 16 + (hi ? 8 : 0);
            float bv = (n < VE) ? bias[n] : 0.0f;
#pragma unroll
            for (int r = 0; r < 8; ++r) {
                int m = mBase + r;
                float y = acc[r] + bv;
                y = (y >= 0.0f) ? y : NSLOPE * y;
                float res = (float)xpad[(size_t)(m + 1) * KP + n];  // x[m][n]
                float o = (n < VE) ? (y + res) : 0.0f;
                out[(size_t)m * KP + n] = (_Float16)o;
            }
        }
    }
}

// ---------------- 5. cosine GEMM: sim = (A*B^T) * invA * invB ---------------
// 2x2 register blocking: each wave produces a 32x32 tile of sim.
__global__ __launch_bounds__(32)
void cosine_kernel(const _Float16* __restrict__ A,   // [QN][KP]
                   const _Float16* __restrict__ B,   // [DN][KP]
                   const float* __restrict__ invA,
                   const float* __restrict__ invB,
                   float* __restrict__ sim) {        // [QN][DN]
    int m0 = blockIdx.x * 32;
    int n0 = blockIdx.y * 32;
    int lane = threadIdx.x;
    int mr = lane & 15;
    bool hi = lane >= 16;
    const _Float16* Arow0 = A + (size_t)(m0 + mr) * KP;
    const _Float16* Arow1 = Arow0 + (size_t)16 * KP;
    const _Float16* Brow0 = B + (size_t)(n0 + mr) * KP;
    const _Float16* Brow1 = Brow0 + (size_t)16 * KP;

    v8f acc00 = {}, acc01 = {}, acc10 = {}, acc11 = {};
#pragma unroll
    for (int k0 = 0; k0 < KP; k0 += 32) {
        v16h a0 = load_a_frag(Arow0, k0, hi);
        v16h a1 = load_a_frag(Arow1, k0, hi);
        v16h b0 = load_b_frag(Brow0, k0, hi);
        v16h b1 = load_b_frag(Brow1, k0, hi);
        acc00 = __builtin_amdgcn_wmma_f32_16x16x32_f16(false, a0, false, b0, (short)0, acc00, false, false);
        acc01 = __builtin_amdgcn_wmma_f32_16x16x32_f16(false, a0, false, b1, (short)0, acc01, false, false);
        acc10 = __builtin_amdgcn_wmma_f32_16x16x32_f16(false, a1, false, b0, (short)0, acc10, false, false);
        acc11 = __builtin_amdgcn_wmma_f32_16x16x32_f16(false, a1, false, b1, (short)0, acc11, false, false);
    }

#pragma unroll
    for (int ti = 0; ti < 2; ++ti) {
#pragma unroll
        for (int tj = 0; tj < 2; ++tj) {
            const v8f& acc = (ti == 0) ? ((tj == 0) ? acc00 : acc01)
                                       : ((tj == 0) ? acc10 : acc11);
            int n = n0 + tj * 16 + mr;
            int mBase = m0 + ti * 16 + (hi ? 8 : 0);
            float ib = invB[n];
#pragma unroll
            for (int r = 0; r < 8; ++r) {
                int m = mBase + r;
                sim[(size_t)m * DN + n] = acc[r] * invA[m] * ib;
            }
        }
    }
}

// ---------------- 6. top-5 pooling -> (max, mean of top5) -------------------
__device__ __forceinline__ void top5_insert(float (&t)[5], float v) {
    if (v <= t[4]) return;
    t[4] = v;
#pragma unroll
    for (int i = 4; i > 0; --i) {
        if (t[i] > t[i - 1]) { float tmp = t[i]; t[i] = t[i - 1]; t[i - 1] = tmp; }
    }
}

// grid: (QN, 6). source s: 0=oh_d1 1=ins_d1 2=sen_d1 3=oh_d2 4=ins_d2 5=sen_d2
__global__ void pool_kernel(const float* __restrict__ oh1,   // doc1_sim [DN][QN]
                            const float* __restrict__ oh2,
                            const float* __restrict__ ins1,  // [QN][DN]
                            const float* __restrict__ ins2,
                            const float* __restrict__ sen1,
                            const float* __restrict__ sen2,
                            float* __restrict__ pooled) {    // [2][QN][6]
    __shared__ float sm[32 * 5];
    int q = blockIdx.x;
    int s = blockIdx.y;
    int lane = threadIdx.x;
    int doc = s / 3, kind = s % 3;

    float t[5] = {-3.0e38f, -3.0e38f, -3.0e38f, -3.0e38f, -3.0e38f};
    if (kind == 0) {
        const float* src = (doc == 0) ? oh1 : oh2;
        for (int l = lane; l < DN; l += 32) top5_insert(t, src[(size_t)l * QN + q]);
    } else {
        const float* src = (kind == 1) ? ((doc == 0) ? ins1 : ins2)
                                       : ((doc == 0) ? sen1 : sen2);
        const float* row = src + (size_t)q * DN;
        for (int l = lane; l < DN; l += 32) top5_insert(t, row[l]);
    }
#pragma unroll
    for (int i = 0; i < 5; ++i) sm[lane * 5 + i] = t[i];
    __syncthreads();
    if (lane == 0) {
        float g[5] = {-3.0e38f, -3.0e38f, -3.0e38f, -3.0e38f, -3.0e38f};
        for (int i = 0; i < 160; ++i) top5_insert(g, sm[i]);
        float mx = g[0];
        float mean = (g[0] + g[1] + g[2] + g[3] + g[4]) * 0.2f;
        float* dst = pooled + ((size_t)doc * QN + q) * 6 + kind * 2;
        dst[0] = mx;
        dst[1] = mean;
    }
}

// ---------------- 7. q weights: qw = q_conv.qw_w[:E] + idf*qw_w[E] + b ------
__global__ void qweights_kernel(const _Float16* __restrict__ qc,  // [QN][KP]
                                const int* __restrict__ question,
                                const float* __restrict__ idf,    // [V]
                                const float* __restrict__ qw_w,   // [E+1]
                                const float* __restrict__ qw_b,   // [1]
                                float* __restrict__ qweights) {
    __shared__ float sv[QN];
    __shared__ float mxs, sms;
    int q = threadIdx.x;
    float s = qw_b[0] + idf[question[q]] * qw_w[VE];
    const _Float16* row = qc + (size_t)q * KP;
    for (int i = 0; i < VE; ++i) s += (float)row[i] * qw_w[i];
    sv[q] = s;
    __syncthreads();
    if (q == 0) {
        float mx = sv[0];
        for (int i = 1; i < QN; ++i) mx = fmaxf(mx, sv[i]);
        float sum = 0.0f;
        for (int i = 0; i < QN; ++i) sum += expf(sv[i] - mx);
        mxs = mx; sms = sum;
    }
    __syncthreads();
    qweights[q] = expf(sv[q] - mxs) / sms;
}

// ---------------- 8. MLP head + hinge loss ----------------------------------
__global__ void final_kernel(const float* __restrict__ pooled,   // [2][QN][6]
                             const float* __restrict__ qweights, // [QN]
                             const float* __restrict__ gaf,      // [4]
                             const float* __restrict__ baf,      // [4]
                             const float* __restrict__ lin1_w,   // [8][6]
                             const float* __restrict__ lin2_w,   // [8]
                             const float* __restrict__ out_w,    // [5]
                             float* __restrict__ d_out) {        // [5]
    __shared__ float r1[QN], r2[QN];
    int q = threadIdx.x;
    float qw = qweights[q];
#pragma unroll
    for (int doc = 0; doc < 2; ++doc) {
        const float* tp = pooled + ((size_t)doc * QN + q) * 6;
        float lo = 0.0f;
#pragma unroll
        for (int j = 0; j < 8; ++j) {
            float h = 0.0f;
#pragma unroll
            for (int i = 0; i < 6; ++i) h += lin1_w[j * 6 + i] * tp[i];
            h = (h >= 0.0f) ? h : NSLOPE * h;
            lo += lin2_w[j] * h;
        }
        if (doc == 0) r1[q] = lo * qw; else r2[q] = lo * qw;
    }
    __syncthreads();
    if (q == 0) {
        float s1 = 0.0f, s2 = 0.0f;
        for (int i = 0; i < QN; ++i) { s1 += r1[i]; s2 += r2[i]; }
        float e1 = s1 / (float)QN, e2 = s2 / (float)QN;
        float good = out_w[4] * e1, bad = out_w[4] * e2;
        for (int i = 0; i < 4; ++i) { good += out_w[i] * gaf[i]; bad += out_w[i] * baf[i]; }
        float loss = fmaxf(0.0f, 1.0f - (good - bad));
        d_out[0] = loss; d_out[1] = good; d_out[2] = bad; d_out[3] = loss; d_out[4] = loss;
    }
}

// ---------------------------------------------------------------------------
extern "C" void kernel_launch(void* const* d_in, const int* in_sizes, int n_in,
                              void* d_out, int out_size, void* d_ws, size_t ws_size,
                              hipStream_t stream) {
    const int*   question = (const int*)  d_in[0];
    const int*   doc1     = (const int*)  d_in[1];
    const int*   doc2     = (const int*)  d_in[2];
    const float* doc1_sim = (const float*)d_in[3];
    const float* doc2_sim = (const float*)d_in[4];
    const float* gaf      = (const float*)d_in[5];
    const float* baf      = (const float*)d_in[6];
    const float* emb      = (const float*)d_in[7];
    const float* idf      = (const float*)d_in[8];
    const float* conv_w   = (const float*)d_in[9];
    const float* conv_b   = (const float*)d_in[10];
    const float* qw_w     = (const float*)d_in[11];
    const float* qw_b     = (const float*)d_in[12];
    const float* lin1_w   = (const float*)d_in[13];
    const float* lin2_w   = (const float*)d_in[14];
    const float* out_w    = (const float*)d_in[15];

    // ---- workspace layout (all offsets 256B aligned) ----
    char* base = (char*)d_ws;
    size_t off = 0;
    auto alloc = [&](size_t bytes) -> char* {
        char* p = base + off;
        off += (bytes + 255) & ~(size_t)255;
        return p;
    };
    _Float16* qpad  = (_Float16*)alloc((size_t)(QN + 2) * KP * 2);
    _Float16* d1pad = (_Float16*)alloc((size_t)(DN + 2) * KP * 2);
    _Float16* d2pad = (_Float16*)alloc((size_t)(DN + 2) * KP * 2);
    _Float16* w0    = (_Float16*)alloc((size_t)KP * KP * 2);
    _Float16* w1    = (_Float16*)alloc((size_t)KP * KP * 2);
    _Float16* w2    = (_Float16*)alloc((size_t)KP * KP * 2);
    _Float16* qc    = (_Float16*)alloc((size_t)QN * KP * 2);
    _Float16* d1c   = (_Float16*)alloc((size_t)DN * KP * 2);
    _Float16* d2c   = (_Float16*)alloc((size_t)DN * KP * 2);
    float* inv_q_raw  = (float*)alloc(QN * 4);
    float* inv_q_conv = (float*)alloc(QN * 4);
    float* inv_d1_raw  = (float*)alloc(DN * 4);
    float* inv_d1_conv = (float*)alloc(DN * 4);
    float* inv_d2_raw  = (float*)alloc(DN * 4);
    float* inv_d2_conv = (float*)alloc(DN * 4);
    float* sim_ins1 = (float*)alloc((size_t)QN * DN * 4);
    float* sim_ins2 = (float*)alloc((size_t)QN * DN * 4);
    float* sim_sen1 = (float*)alloc((size_t)QN * DN * 4);
    float* sim_sen2 = (float*)alloc((size_t)QN * DN * 4);
    float* pooled   = (float*)alloc(2 * QN * 6 * 4);
    float* qweights = (float*)alloc(QN * 4);
    (void)ws_size; (void)n_in; (void)in_sizes; (void)out_size;

    _Float16* qraw  = qpad  + KP;   // skip halo row 0
    _Float16* d1raw = d1pad + KP;
    _Float16* d2raw = d2pad + KP;

    // 1) gather + convert (padded)
    gather_kernel<<<(QN + 2) + 2 * (DN + 2), KP, 0, stream>>>(
        question, doc1, doc2, emb, qpad, d1pad, d2pad);

    // 2) conv tap matrices
    taps_kernel<<<3 * KP, KP, 0, stream>>>(conv_w, w0, w1, w2);

    // 3) raw norms
    norm_kernel<<<(QN + 3) / 4, 128, 0, stream>>>(qraw,  QN, inv_q_raw);
    norm_kernel<<<(DN + 3) / 4, 128, 0, stream>>>(d1raw, DN, inv_d1_raw);
    norm_kernel<<<(DN + 3) / 4, 128, 0, stream>>>(d2raw, DN, inv_d2_raw);

    // 4) conv residual (WMMA, 32x32 tiles)
    conv_kernel<<<dim3(QN / 32, KP / 32), 32, 0, stream>>>(qpad,  w0, w1, w2, conv_b, qc,  QN);
    conv_kernel<<<dim3(DN / 32, KP / 32), 32, 0, stream>>>(d1pad, w0, w1, w2, conv_b, d1c, DN);
    conv_kernel<<<dim3(DN / 32, KP / 32), 32, 0, stream>>>(d2pad, w0, w1, w2, conv_b, d2c, DN);

    // 5) conv norms
    norm_kernel<<<(QN + 3) / 4, 128, 0, stream>>>(qc,  QN, inv_q_conv);
    norm_kernel<<<(DN + 3) / 4, 128, 0, stream>>>(d1c, DN, inv_d1_conv);
    norm_kernel<<<(DN + 3) / 4, 128, 0, stream>>>(d2c, DN, inv_d2_conv);

    // 6) cosine similarity GEMMs (WMMA, 32x32 tiles), normalization fused
    cosine_kernel<<<dim3(QN / 32, DN / 32), 32, 0, stream>>>(qraw, d1raw, inv_q_raw,  inv_d1_raw,  sim_ins1);
    cosine_kernel<<<dim3(QN / 32, DN / 32), 32, 0, stream>>>(qraw, d2raw, inv_q_raw,  inv_d2_raw,  sim_ins2);
    cosine_kernel<<<dim3(QN / 32, DN / 32), 32, 0, stream>>>(qc,   d1c,   inv_q_conv, inv_d1_conv, sim_sen1);
    cosine_kernel<<<dim3(QN / 32, DN / 32), 32, 0, stream>>>(qc,   d2c,   inv_q_conv, inv_d2_conv, sim_sen2);

    // 7) top-5 pooling over 6 sources
    pool_kernel<<<dim3(QN, 6), 32, 0, stream>>>(doc1_sim, doc2_sim,
                                                sim_ins1, sim_ins2,
                                                sim_sen1, sim_sen2, pooled);

    // 8) query weights (softmax)
    qweights_kernel<<<1, QN, 0, stream>>>(qc, question, idf, qw_w, qw_b, qweights);

    // 9) head + hinge loss
    final_kernel<<<1, QN, 0, stream>>>(pooled, qweights, gaf, baf,
                                       lin1_w, lin2_w, out_w, (float*)d_out);
}
